// NystromSelfAttention_15418932593418
// MI455X (gfx1250) — matrix operand order repaired and checked by reading, not compiled, not run on hardware
//
#include <hip/hip_runtime.h>

// ---------------------------------------------------------------------------
// Types / helpers
// ---------------------------------------------------------------------------
typedef __bf16 bf16;
typedef __attribute__((ext_vector_type(16))) __bf16 v16bf;
typedef __attribute__((ext_vector_type(8)))  float  v8f;
typedef __attribute__((ext_vector_type(4)))  unsigned int uint4v;

union Frag { v16bf v; uint4v u[2]; };

__device__ __forceinline__ bf16 f2bf(float f) {
  unsigned u = __float_as_uint(f);
  unsigned r = (u + 0x7FFFu + ((u >> 16) & 1u)) >> 16;
  unsigned short s = (unsigned short)r;
  return __builtin_bit_cast(bf16, s);
}
__device__ __forceinline__ float bf2f(bf16 h) {
  unsigned short s = __builtin_bit_cast(unsigned short, h);
  return __uint_as_float(((unsigned)s) << 16);
}
__device__ __forceinline__ v8f vzero8() {
  v8f z = {0.f,0.f,0.f,0.f,0.f,0.f,0.f,0.f};
  return z;
}
// A-fragment (16x32 bf16): lane L reads k = kb..kb+7 and kb+16..kb+23, kb = 8*(L/16)
__device__ __forceinline__ v16bf frag_a(const bf16* p, int lane) {
  p += (lane >> 4) << 3;
  Frag f;
  f.u[0] = *(const uint4v*)p;
  f.u[1] = *(const uint4v*)(p + 16);
  return f.v;
}
// B-fragment (32x16 bf16): lane L (col = L%16) reads k = kb..kb+7 and kb+16..kb+23, kb = 8*(L/16)
__device__ __forceinline__ v16bf frag_b(const bf16* p, int lane) {
  p += (lane >> 4) << 3;
  Frag f;
  f.u[0] = *(const uint4v*)p;
  f.u[1] = *(const uint4v*)(p + 16);
  return f.v;
}
__device__ __forceinline__ v8f wmma_bf16(v16bf a, v16bf b, v8f c) {
  return __builtin_amdgcn_wmma_f32_16x16x32_bf16(false, a, false, b, (short)0, c,
                                                 false, false);
}

#define Bsz 4
#define Nsq 4096
#define Csz 1024
#define Hh  16
#define Mlm 64
#define DH  64
#define BH  (Bsz*Hh)
#define ROWS (Bsz*Nsq)

// ---------------------------------------------------------------------------
// f32 -> bf16 convert
// ---------------------------------------------------------------------------
__global__ __launch_bounds__(256) void cvt_bf16(const float* __restrict__ src,
                                                bf16* __restrict__ dst, int n) {
  int i = blockIdx.x * 256 + threadIdx.x;
  if (i < n) dst[i] = f2bf(src[i]);
}

// ---------------------------------------------------------------------------
// LayerNorm over C=1024, output bf16
// ---------------------------------------------------------------------------
__global__ __launch_bounds__(256) void layernorm_k(const float* __restrict__ X,
    const float* __restrict__ w, const float* __restrict__ b,
    bf16* __restrict__ O) {
  __shared__ float red[256];
  const size_t row = blockIdx.x;
  const float* p = X + row * Csz;
  float v[4]; float s = 0.f;
  for (int j = 0; j < 4; ++j) { v[j] = p[threadIdx.x + j*256]; s += v[j]; }
  red[threadIdx.x] = s; __syncthreads();
  for (int st = 128; st > 0; st >>= 1) {
    if (threadIdx.x < st) red[threadIdx.x] += red[threadIdx.x + st];
    __syncthreads();
  }
  float mu = red[0] * (1.f/1024.f); __syncthreads();
  float sq = 0.f;
  for (int j = 0; j < 4; ++j) { float d = v[j] - mu; sq += d*d; }
  red[threadIdx.x] = sq; __syncthreads();
  for (int st = 128; st > 0; st >>= 1) {
    if (threadIdx.x < st) red[threadIdx.x] += red[threadIdx.x + st];
    __syncthreads();
  }
  float rstd = rsqrtf(red[0] * (1.f/1024.f) + 1e-5f);
  for (int j = 0; j < 4; ++j) {
    int c = threadIdx.x + j*256;
    O[row*Csz + c] = f2bf((v[j]-mu)*rstd*w[c] + b[c]);
  }
}

// ---------------------------------------------------------------------------
// Generic big WMMA GEMM: D[row,col] = sum_k A[row,k]*W[col,k] + bias[col]
// 128x128 block tile, double-buffered LDS, 8 waves of 64x32 wave tiles.
// Global loads for K-step k+1 are issued before the WMMAs of step k; the
// register->LDS stores land in the alternate buffer before the one barrier.
// __builtin_prefetch pulls step k+2 toward cache (global_prefetch_b8).
// MODE 0: qkv scatter to q/k/v [B,H,N,DH] bf16 (+ vT [B,H,DH,N])
// MODE 1: bf16 row-major (stride 1024)
// MODE 2: f32  row-major (stride 1024)
// ---------------------------------------------------------------------------
template<int MODE>
__global__ __launch_bounds__(256) void gemm128_wmma(
    const bf16* __restrict__ A, const bf16* __restrict__ W,
    const float* __restrict__ bias,
    bf16* __restrict__ o0, bf16* __restrict__ o1,
    bf16* __restrict__ o2, bf16* __restrict__ o3,
    float* __restrict__ of, int K) {
  __shared__ bf16 As[2][128][48];
  __shared__ bf16 Ws[2][128][48];
  const int t = threadIdx.x;
  const int row0 = blockIdx.y * 128;
  const int col0 = blockIdx.x * 128;
  const int wave = t >> 5, lane = t & 31;
  const int wm = wave >> 2, wn = wave & 3;
  const int lrow = lane & 15;
  // per-thread staging slots: 2 chunks of 8 bf16 for each of A / W
  const int sr0 = t >> 2,          sc0 = (t & 3) << 3;          // chunk 0
  const int sr1 = (256 + t) >> 2,  sc1 = (t & 3) << 3;          // chunk 1
  v8f acc[4][2];
  for (int mt = 0; mt < 4; ++mt)
    for (int nt = 0; nt < 2; ++nt) acc[mt][nt] = vzero8();

  uint4v ra0, ra1, rw0, rw1;
  // preload K-step 0 into registers, store to buffer 0
  ra0 = *(const uint4v*)&A[(size_t)(row0 + sr0)*K + sc0];
  ra1 = *(const uint4v*)&A[(size_t)(row0 + sr1)*K + sc1];
  rw0 = *(const uint4v*)&W[(size_t)(col0 + sr0)*K + sc0];
  rw1 = *(const uint4v*)&W[(size_t)(col0 + sr1)*K + sc1];
  *(uint4v*)&As[0][sr0][sc0] = ra0;
  *(uint4v*)&As[0][sr1][sc1] = ra1;
  *(uint4v*)&Ws[0][sr0][sc0] = rw0;
  *(uint4v*)&Ws[0][sr1][sc1] = rw1;
  __syncthreads();

  for (int k0 = 0; k0 < K; k0 += 32) {
    const int cur = (k0 >> 5) & 1;
    const bool more = (k0 + 32) < K;
    if (more) {                      // issue next tile's global loads early
      ra0 = *(const uint4v*)&A[(size_t)(row0 + sr0)*K + k0 + 32 + sc0];
      ra1 = *(const uint4v*)&A[(size_t)(row0 + sr1)*K + k0 + 32 + sc1];
      rw0 = *(const uint4v*)&W[(size_t)(col0 + sr0)*K + k0 + 32 + sc0];
      rw1 = *(const uint4v*)&W[(size_t)(col0 + sr1)*K + k0 + 32 + sc1];
    }
    if (k0 + 64 < K) {               // prefetch step k+2 toward cache
      __builtin_prefetch(&A[(size_t)(row0 + sr0)*K + k0 + 64 + sc0], 0, 0);
      __builtin_prefetch(&W[(size_t)(col0 + sr0)*K + k0 + 64 + sc0], 0, 0);
    }
    v16bf af[4], bfr[2];
    for (int mt = 0; mt < 4; ++mt) af[mt]  = frag_a(&As[cur][wm*64 + mt*16 + lrow][0], lane);
    for (int nt = 0; nt < 2; ++nt) bfr[nt] = frag_b(&Ws[cur][wn*32 + nt*16 + lrow][0], lane);
    for (int mt = 0; mt < 4; ++mt)
      for (int nt = 0; nt < 2; ++nt)
        acc[mt][nt] = wmma_bf16(af[mt], bfr[nt], acc[mt][nt]);
    if (more) {                      // commit next tile to alternate buffer
      const int nxt = cur ^ 1;
      *(uint4v*)&As[nxt][sr0][sc0] = ra0;
      *(uint4v*)&As[nxt][sr1][sc1] = ra1;
      *(uint4v*)&Ws[nxt][sr0][sc0] = rw0;
      *(uint4v*)&Ws[nxt][sr1][sc1] = rw1;
    }
    __syncthreads();
  }

  for (int mt = 0; mt < 4; ++mt)
    for (int nt = 0; nt < 2; ++nt)
      for (int r = 0; r < 8; ++r) {
        int rr = row0 + wm*64 + mt*16 + r + ((lane >> 4) << 3);
        int cc = col0 + wn*32 + nt*16 + (lane & 15);
        float val = acc[mt][nt][r] + bias[cc];
        if (MODE == 0) {
          int which = cc >> 10, rem = cc & 1023;
          int h = rem >> 6, d = rem & 63;
          int b = rr >> 12, n = rr & 4095;
          size_t o = ((((size_t)b << 4) + h) * Nsq + n) * DH + d;
          bf16 bv = f2bf(val);
          if (which == 0) o0[o] = bv;
          else if (which == 1) o1[o] = bv;
          else {
            o2[o] = bv;
            o3[((((size_t)b << 4) + h) * DH + d) * Nsq + n] = bv;
          }
        } else if (MODE == 1) {
          o0[(size_t)rr * Csz + cc] = f2bf(val);
        } else {
          of[(size_t)rr * Csz + cc] = val;
        }
      }
}

// ---------------------------------------------------------------------------
// Landmark mean-pool: [BH,4096,64] bf16 -> [BH,64,64] (f32 + bf16), * SCALE/64
// ---------------------------------------------------------------------------
__global__ __launch_bounds__(256) void pool_lm(const bf16* __restrict__ X,
    float* __restrict__ Lf, bf16* __restrict__ Lb) {
  int idx = blockIdx.x * 256 + threadIdx.x;   // BH*64*64
  int d = idx & 63, m = (idx >> 6) & 63, bh = idx >> 12;
  const bf16* p = X + ((size_t)bh * Nsq + m * 64) * DH + d;
  float s = 0.f;
  for (int j = 0; j < 64; ++j) s += bf2f(p[(size_t)j * DH]);
  s *= (0.25f / 64.f);                        // mean * SCALE (H^-0.5 = 0.25)
  Lf[idx] = s;
  Lb[idx] = f2bf(s);
}

// ---------------------------------------------------------------------------
// [rows x 64] = A[bh] (rows x 64) @ W[bh]^T (64 x 64), optional row-softmax.
// SCATTER: write to [B,N,C] with head offset. Otherwise [BH,rows,64] bf16.
// Block: 128 rows x 64 cols; 8 waves each one 16x64 strip (4 WMMA tiles).
// ---------------------------------------------------------------------------
template<bool SOFT, bool SCATTER>
__global__ __launch_bounds__(256) void attn_nx64(
    const bf16* __restrict__ A, const bf16* __restrict__ W,
    bf16* __restrict__ out, int rows) {
  const int bh = blockIdx.y;
  const int row0 = blockIdx.x * 128;
  const int t = threadIdx.x, wave = t >> 5, lane = t & 31;
  const int lrow = lane & 15;
  const bf16* pa = A + ((size_t)bh * rows + row0 + wave*16 + lrow) * 64;
  const bf16* pw = W + ((size_t)bh * 64 + lrow) * 64;
  v8f acc[4];
  for (int nt = 0; nt < 4; ++nt) acc[nt] = vzero8();
  #pragma unroll
  for (int ks = 0; ks < 64; ks += 32) {
    v16bf af = frag_a(pa + ks, lane);
    #pragma unroll
    for (int nt = 0; nt < 4; ++nt) {
      v16bf bfr = frag_b(pw + (size_t)nt * (16*64) + ks, lane);
      acc[nt] = wmma_bf16(af, bfr, acc[nt]);
    }
  }
  if (SOFT) {
    for (int r = 0; r < 8; ++r) {
      float v0 = acc[0][r], v1 = acc[1][r], v2 = acc[2][r], v3 = acc[3][r];
      float mx = fmaxf(fmaxf(v0, v1), fmaxf(v2, v3));
      for (int m = 1; m < 16; m <<= 1) mx = fmaxf(mx, __shfl_xor(mx, m, 32));
      v0 = __expf(v0 - mx); v1 = __expf(v1 - mx);
      v2 = __expf(v2 - mx); v3 = __expf(v3 - mx);
      float s = v0 + v1 + v2 + v3;
      for (int m = 1; m < 16; m <<= 1) s += __shfl_xor(s, m, 32);
      float inv = 1.f / s;
      acc[0][r] = v0*inv; acc[1][r] = v1*inv; acc[2][r] = v2*inv; acc[3][r] = v3*inv;
    }
  }
  for (int nt = 0; nt < 4; ++nt)
    for (int r = 0; r < 8; ++r) {
      int rr = row0 + wave*16 + r + ((lane >> 4) << 3);
      int cc = nt*16 + (lane & 15);
      bf16 bv = f2bf(acc[nt][r]);
      if (SCATTER) {
        int b = bh >> 4, h = bh & 15;
        out[((size_t)b * Nsq + rr) * Csz + h*64 + cc] = bv;
      } else {
        out[((size_t)bh * rows + rr) * 64 + cc] = bv;
      }
    }
}

// ---------------------------------------------------------------------------
// attn3 scores: S[bh][m][n] = q_lm[bh][m,:] . k[bh][n,:]   (f32 out)
// Block: 64 rows x 128 cols; 8 waves (4 m-tiles x 2 col-halves of 64).
// ---------------------------------------------------------------------------
__global__ __launch_bounds__(256) void lm_scores(
    const bf16* __restrict__ Qlm, const bf16* __restrict__ Kb,
    float* __restrict__ S) {
  const int bh = blockIdx.y;
  const int col0 = blockIdx.x * 128;
  const int t = threadIdx.x, wave = t >> 5, lane = t & 31;
  const int wm = wave & 3, wn2 = wave >> 2;
  const int lrow = lane & 15;
  const bf16* pa = Qlm + ((size_t)bh * 64 + wm*16 + lrow) * 64;
  const bf16* pw = Kb + ((size_t)bh * Nsq + col0 + wn2*64 + lrow) * 64;
  v8f acc[4];
  for (int nt = 0; nt < 4; ++nt) acc[nt] = vzero8();
  #pragma unroll
  for (int ks = 0; ks < 64; ks += 32) {
    v16bf af = frag_a(pa + ks, lane);
    #pragma unroll
    for (int nt = 0; nt < 4; ++nt) {
      v16bf bfr = frag_b(pw + (size_t)nt * (16*64) + ks, lane);
      acc[nt] = wmma_bf16(af, bfr, acc[nt]);
    }
  }
  for (int nt = 0; nt < 4; ++nt)
    for (int r = 0; r < 8; ++r) {
      int m = wm*16 + r + ((lane >> 4) << 3);
      int n = col0 + wn2*64 + nt*16 + (lane & 15);
      S[((size_t)bh * 64 + m) * Nsq + n] = acc[nt][r];
    }
}

// ---------------------------------------------------------------------------
// Row softmax over length 4096 (f32 in, bf16 out)
// ---------------------------------------------------------------------------
__global__ __launch_bounds__(256) void softmax_row4096(
    const float* __restrict__ S, bf16* __restrict__ O) {
  __shared__ float red[256];
  const size_t row = blockIdx.x;
  const float* p = S + row * Nsq;
  float mx = -3.4e38f;
  for (int i = threadIdx.x; i < Nsq; i += 256) mx = fmaxf(mx, p[i]);
  red[threadIdx.x] = mx; __syncthreads();
  for (int st = 128; st > 0; st >>= 1) {
    if (threadIdx.x < st) red[threadIdx.x] = fmaxf(red[threadIdx.x], red[threadIdx.x + st]);
    __syncthreads();
  }
  mx = red[0]; __syncthreads();
  float s = 0.f;
  for (int i = threadIdx.x; i < Nsq; i += 256) s += __expf(p[i] - mx);
  red[threadIdx.x] = s; __syncthreads();
  for (int st = 128; st > 0; st >>= 1) {
    if (threadIdx.x < st) red[threadIdx.x] += red[threadIdx.x + st];
    __syncthreads();
  }
  float inv = 1.f / red[0];
  for (int i = threadIdx.x; i < Nsq; i += 256)
    O[row * Nsq + i] = f2bf(__expf(p[i] - mx) * inv);
}

// ---------------------------------------------------------------------------
// attn2 softmax + Newton-Schulz pseudo-inverse, one block per (b,h), f32 LDS
// ---------------------------------------------------------------------------
__device__ __forceinline__ void mm64(float (&D)[64][64], const float (&Aa)[64][64],
                                     const float (&Bb)[64][64], int t) {
  for (int i = t; i < 4096; i += 256) {
    int m = i >> 6, n = i & 63;
    float s = 0.f;
    for (int k = 0; k < 64; ++k) s = fmaf(Aa[m][k], Bb[k][n], s);
    D[m][n] = s;
  }
}

__global__ __launch_bounds__(256) void attn2_ns(
    const float* __restrict__ Qlm, const float* __restrict__ Klm,
    float* __restrict__ INV) {
  __shared__ float Km[64][64], Vm[64][64], KV[64][64], Tm[64][64], Um[64][64];
  __shared__ float cs[64], rs[64], scal;
  const int bh = blockIdx.x, t = threadIdx.x;
  const float* Q  = Qlm + (size_t)bh * 4096;
  const float* Kp = Klm + (size_t)bh * 4096;
  for (int i = t; i < 4096; i += 256) {
    int m = i >> 6, l = i & 63;
    float s = 0.f;
    for (int d = 0; d < 64; ++d) s = fmaf(Q[m*64 + d], Kp[l*64 + d], s);
    Km[m][l] = s;
  }
  __syncthreads();
  if (t < 64) {                               // row softmax
    float mx = -3.4e38f;
    for (int j = 0; j < 64; ++j) mx = fmaxf(mx, Km[t][j]);
    float sm = 0.f;
    for (int j = 0; j < 64; ++j) { float e = __expf(Km[t][j] - mx); Km[t][j] = e; sm += e; }
    float inv = 1.f / sm;
    for (int j = 0; j < 64; ++j) Km[t][j] *= inv;
  }
  __syncthreads();
  if (t < 64) {
    float c = 0.f, r = 0.f;
    for (int j = 0; j < 64; ++j) { c += fabsf(Km[j][t]); r += fabsf(Km[t][j]); }
    cs[t] = c; rs[t] = r;
  }
  __syncthreads();
  if (t == 0) {
    float v0 = 0.f, vi = 0.f;
    for (int j = 0; j < 64; ++j) { v0 = fmaxf(v0, cs[j]); vi = fmaxf(vi, rs[j]); }
    scal = 1.f / (v0 * vi);
  }
  __syncthreads();
  for (int i = t; i < 4096; i += 256) { int m = i >> 6, l = i & 63; Vm[m][l] = Km[l][m] * scal; }
  __syncthreads();
  for (int it = 0; it < 6; ++it) {
    mm64(KV, Km, Vm, t); __syncthreads();
    for (int i = t; i < 4096; i += 256) { int m = i >> 6, l = i & 63; Tm[m][l] = (m == l ? 7.f : 0.f) - KV[m][l]; }
    __syncthreads();
    mm64(Um, KV, Tm, t); __syncthreads();
    for (int i = t; i < 4096; i += 256) { int m = i >> 6, l = i & 63; Tm[m][l] = (m == l ? 15.f : 0.f) - Um[m][l]; }
    __syncthreads();
    mm64(Um, KV, Tm, t); __syncthreads();
    for (int i = t; i < 4096; i += 256) { int m = i >> 6, l = i & 63; Tm[m][l] = (m == l ? 13.f : 0.f) - Um[m][l]; }
    __syncthreads();
    mm64(Um, Vm, Tm, t); __syncthreads();
    for (int i = t; i < 4096; i += 256) { int m = i >> 6, l = i & 63; Vm[m][l] = 0.25f * Um[m][l]; }
    __syncthreads();
  }
  for (int i = t; i < 4096; i += 256) INV[(size_t)bh * 4096 + i] = Vm[i >> 6][i & 63];
}

// ---------------------------------------------------------------------------
// t1[bh] = attn3[bh] (64x4096) @ vT[bh]^T  -> 64x64 f32, K = 4096 WMMA loop
// 8 waves: 4 m-tiles x 2 col-halves (wave tile 16x32)
// ---------------------------------------------------------------------------
__global__ __launch_bounds__(256) void t1_gemm(
    const bf16* __restrict__ A3, const bf16* __restrict__ VT,
    float* __restrict__ T1) {
  const int bh = blockIdx.x;
  const int t = threadIdx.x, wave = t >> 5, lane = t & 31;
  const int wm = wave >> 1, wn = wave & 1;
  const int lrow = lane & 15;
  const bf16* pa  = A3 + ((size_t)bh * 64 + wm*16 + lrow) * Nsq;
  const bf16* pw0 = VT + ((size_t)bh * 64 + wn*32 + lrow) * Nsq;
  const bf16* pw1 = pw0 + (size_t)16 * Nsq;
  v8f acc[2];
  acc[0] = vzero8(); acc[1] = vzero8();
  for (int k0 = 0; k0 < Nsq; k0 += 32) {
    v16bf af = frag_a(pa + k0, lane);
    acc[0] = wmma_bf16(af, frag_b(pw0 + k0, lane), acc[0]);
    acc[1] = wmma_bf16(af, frag_b(pw1 + k0, lane), acc[1]);
  }
  for (int nt = 0; nt < 2; ++nt)
    for (int r = 0; r < 8; ++r) {
      int m = wm*16 + r + ((lane >> 4) << 3);
      int d = wn*32 + nt*16 + (lane & 15);
      T1[((size_t)bh * 64 + m) * 64 + d] = acc[nt][r];
    }
}

// ---------------------------------------------------------------------------
// t2 = attn2_inv @ t1 (64x64x64 f32), emit transposed bf16 [bh][d][m]
// ---------------------------------------------------------------------------
__global__ __launch_bounds__(256) void t2_gemm(
    const float* __restrict__ INV, const float* __restrict__ T1,
    bf16* __restrict__ T2T) {
  __shared__ float Am[64][64], Bm[64][64];
  const int bh = blockIdx.x, t = threadIdx.x;
  for (int i = t; i < 4096; i += 256) {
    Am[i >> 6][i & 63] = INV[(size_t)bh * 4096 + i];
    Bm[i >> 6][i & 63] = T1[(size_t)bh * 4096 + i];
  }
  __syncthreads();
  for (int i = t; i < 4096; i += 256) {
    int m = i >> 6, d = i & 63;
    float s = 0.f;
    for (int k = 0; k < 64; ++k) s = fmaf(Am[m][k], Bm[k][d], s);
    T2T[((size_t)bh * 64 + d) * 64 + m] = f2bf(s);
  }
}

// ---------------------------------------------------------------------------
// Host launcher
// ---------------------------------------------------------------------------
extern "C" void kernel_launch(void* const* d_in, const int* in_sizes, int n_in,
                              void* d_out, int out_size, void* d_ws, size_t ws_size,
                              hipStream_t stream) {
  const float* x       = (const float*)d_in[0];
  const float* ln_w    = (const float*)d_in[1];
  const float* ln_b    = (const float*)d_in[2];
  const float* qkv_w   = (const float*)d_in[3];
  const float* qkv_b   = (const float*)d_in[4];
  const float* gamma_w = (const float*)d_in[5];
  const float* gamma_b = (const float*)d_in[6];
  const float* proj_w  = (const float*)d_in[7];
  const float* proj_b  = (const float*)d_in[8];
  float* out = (float*)d_out;

  char* w = (char*)d_ws;
  size_t off = 0;
  auto alloc = [&](size_t bytes) -> char* {
    char* p = w + off;
    off = (off + bytes + 255) & ~(size_t)255;
    return p;
  };
  const size_t NHD = (size_t)BH * Nsq * DH;     // 16,777,216
  bf16*  wqkv  = (bf16*) alloc((size_t)3*Csz*Csz * 2);
  bf16*  wg    = (bf16*) alloc((size_t)Csz*Csz * 2);
  bf16*  wp    = (bf16*) alloc((size_t)Csz*Csz * 2);
  bf16*  xn    = (bf16*) alloc((size_t)ROWS*Csz * 2);
  bf16*  qb    = (bf16*) alloc(NHD * 2);
  bf16*  kb_   = (bf16*) alloc(NHD * 2);
  bf16*  vb    = (bf16*) alloc(NHD * 2);
  bf16*  vT    = (bf16*) alloc(NHD * 2);
  float* qlm_f = (float*)alloc((size_t)BH*64*64 * 4);
  float* klm_f = (float*)alloc((size_t)BH*64*64 * 4);
  bf16*  qlm_b = (bf16*) alloc((size_t)BH*64*64 * 2);
  bf16*  klm_b = (bf16*) alloc((size_t)BH*64*64 * 2);
  bf16*  attn1 = (bf16*) alloc(NHD * 2);
  float* sc3   = (float*)alloc((size_t)BH*64*Nsq * 4);
  bf16*  attn3 = (bf16*) alloc((size_t)BH*64*Nsq * 2);
  float* invb  = (float*)alloc((size_t)BH*64*64 * 4);
  float* t1    = (float*)alloc((size_t)BH*64*64 * 4);
  bf16*  t2T   = (bf16*) alloc((size_t)BH*64*64 * 2);
  bf16*  ao    = (bf16*) alloc((size_t)ROWS*Csz * 2);
  bf16*  y1    = (bf16*) alloc((size_t)ROWS*Csz * 2);
  (void)ws_size; (void)in_sizes; (void)n_in; (void)out_size;

  // 1) weight converts + layernorm
  cvt_bf16<<<(3*Csz*Csz)/256, 256, 0, stream>>>(qkv_w,   wqkv, 3*Csz*Csz);
  cvt_bf16<<<(Csz*Csz)/256,   256, 0, stream>>>(gamma_w, wg,   Csz*Csz);
  cvt_bf16<<<(Csz*Csz)/256,   256, 0, stream>>>(proj_w,  wp,   Csz*Csz);
  layernorm_k<<<ROWS, 256, 0, stream>>>(x, ln_w, ln_b, xn);

  // 2) QKV GEMM (scatter to q/k/v + vT)
  gemm128_wmma<0><<<dim3((3*Csz)/128, ROWS/128), 256, 0, stream>>>(
      xn, wqkv, qkv_b, qb, kb_, vb, vT, nullptr, Csz);

  // 3) landmarks
  pool_lm<<<(BH*64*64)/256, 256, 0, stream>>>(qb, qlm_f, qlm_b);
  pool_lm<<<(BH*64*64)/256, 256, 0, stream>>>(kb_, klm_f, klm_b);

  // 4) attn1 = softmax(q @ k_lm^T)
  attn_nx64<true, false><<<dim3(Nsq/128, BH), 256, 0, stream>>>(qb, klm_b, attn1, Nsq);

  // 5) attn3 = softmax(q_lm @ k^T) over N
  lm_scores<<<dim3(Nsq/128, BH), 256, 0, stream>>>(qlm_b, kb_, sc3);
  softmax_row4096<<<BH*64, 256, 0, stream>>>(sc3, attn3);

  // 6) attn2 softmax + Newton-Schulz inverse
  attn2_ns<<<BH, 256, 0, stream>>>(qlm_f, klm_f, invb);

  // 7) t1 = attn3 @ v ; t2 = inv @ t1 (emit t2^T)
  t1_gemm<<<BH, 256, 0, stream>>>(attn3, vT, t1);
  t2_gemm<<<BH, 256, 0, stream>>>(invb, t1, t2T);

  // 8) out_heads = attn1 @ t2, scattered to [B,N,C]
  attn_nx64<false, true><<<dim3(Nsq/128, BH), 256, 0, stream>>>(attn1, t2T, ao, Nsq);

  // 9) gamma then proj GEMMs
  gemm128_wmma<1><<<dim3(Csz/128, ROWS/128), 256, 0, stream>>>(
      ao, wg, gamma_b, y1, nullptr, nullptr, nullptr, nullptr, Csz);
  gemm128_wmma<2><<<dim3(Csz/128, ROWS/128), 256, 0, stream>>>(
      y1, wp, proj_b, nullptr, nullptr, nullptr, nullptr, out, Csz);
}
